// MaskRCNN_1142461301041
// MI455X (gfx1250) — compile-verified, matrix operand order checked by reading, not run on hardware
//
#include <hip/hip_runtime.h>

// ---------------------------------------------------------------------------
// Types for CDNA5 WMMA (wave32, v_wmma_f32_16x16x32_bf16)
// ---------------------------------------------------------------------------
typedef __bf16 v16bf __attribute__((ext_vector_type(16)));
typedef __bf16 v8bf  __attribute__((ext_vector_type(8)));
typedef float  v8f   __attribute__((ext_vector_type(8)));

__device__ __forceinline__ float bfbits2f(unsigned short u) {
  union { unsigned u; float f; } v; v.u = ((unsigned)u) << 16; return v.f;
}
__device__ __forceinline__ unsigned short f2bfbits(float f) {
  union { float f; unsigned u; } v; v.f = f;
  unsigned r = v.u + 0x7FFFu + ((v.u >> 16) & 1u);
  return (unsigned short)(r >> 16);
}

// ---------------------------------------------------------------------------
// Prep kernels: f32->bf16 convert, weight reshape [OC][K]->[OC][Kp] (zero pad),
// BN folding  scale = g/sqrt(v+eps), bias = b - m*scale
// ---------------------------------------------------------------------------
__global__ void cvt_f32_bf16_k(const float* __restrict__ src,
                               unsigned short* __restrict__ dst, int n) {
  int i = blockIdx.x * blockDim.x + threadIdx.x;
  if (i < n) dst[i] = f2bfbits(src[i]);
}

__global__ void convert_w_k(const float* __restrict__ src,
                            unsigned short* __restrict__ dst,
                            int OC, int K, int Kp) {
  int i = blockIdx.x * blockDim.x + threadIdx.x;
  if (i >= OC * Kp) return;
  int oc = i / Kp, k = i % Kp;
  dst[i] = (k < K) ? f2bfbits(src[oc * K + k]) : 0;
}

__global__ void fold_bn_k(const float* __restrict__ g, const float* __restrict__ b,
                          const float* __restrict__ m, const float* __restrict__ v,
                          float* __restrict__ scale, float* __restrict__ bias, int C) {
  int c = blockIdx.x * blockDim.x + threadIdx.x;
  if (c >= C) return;
  float s = g[c] * rsqrtf(v[c] + 1e-5f);
  scale[c] = s;
  bias[c]  = b[c] - m[c] * s;
}

// ---------------------------------------------------------------------------
// Implicit-GEMM convolution on WMMA.
//   GEMM: M = Nimg*OH*OW (output pixels), N = OC, K = C*KH*KW
//   One wave computes a 16(M) x 32(N) tile: 2 f32 accumulators sharing one A
//   fragment. K-loop in bf16 chunks of 32, unrolled by 2.
//   A tile: im2col gather staged through LDS (documented 16-bit A layout:
//   lanes 0-15 hold K 0-7/16-23, lanes 16-31 hold K 8-15/24-31, so fragments
//   are two contiguous 16B ds_load_b128 per lane).
//   B tile (weights, contiguous [oc][Kp] rows): staged 32x32 into LDS with
//   GLOBAL_LOAD_ASYNC_TO_LDS_B128 (4 x 16B per lane), synced by ASYNCcnt.
//   Epilogue: per-channel scale/bias (folded BN), optional residual, ReLU.
// ---------------------------------------------------------------------------
__global__ __launch_bounds__(32)
void conv_wmma_k(const unsigned short* __restrict__ in,
                 const unsigned short* __restrict__ wt,
                 const float* __restrict__ scale, const float* __restrict__ bias,
                 const unsigned short* __restrict__ res,
                 unsigned short* __restrict__ out,
                 int Nimg, int C, int H, int W, int OC, int OH, int OW,
                 int KH, int KW, int stride, int pad, int K, int Kp, int relu) {
  __shared__ __align__(16) unsigned short ldsA[16 * 32];
  __shared__ __align__(16) unsigned short ldsB[32 * 32];

  const int lane = threadIdx.x;
  const int m0 = blockIdx.x * 16;
  const int n0 = blockIdx.y * 32;
  const int M = Nimg * OH * OW;
  const int KK = KH * KW;

  // Row of A this lane stages (two lanes per row, one per 16-wide K half).
  const int srow = lane & 15;
  const int khalf = (lane >> 4) * 16;
  const int rowm = m0 + srow;
  const bool rowok = rowm < M;
  int n_img = 0, oh = 0, ow = 0;
  if (rowok) {
    n_img = rowm / (OH * OW);
    int r = rowm % (OH * OW);
    oh = r / OW; ow = r % OW;
  }
  const int ih0 = oh * stride - pad;
  const int iw0 = ow * stride - pad;

  // LDS byte offsets (generic LDS pointer: low 32 bits are the LDS address).
  const unsigned ldsB_off = (unsigned)(size_t)(void*)ldsB;

  // Per-lane async-copy addressing for the B tile (32 rows x 64B).
  const int blin = lane * 16;             // 16B per lane per issue
  const int brow_st = blin >> 6;          // staging row for issue 0
  const int bbyte = blin & 63;
  const unsigned long long wt_base = (unsigned long long)(size_t)wt;

  // B fragment addressing after staging (16-bit B layout mirrors A).
  const int frow = lane & 15;
  const int fk = (lane >> 4) * 8;

  v8f acc0 = 0.0f, acc1 = 0.0f;

  auto chunk = [&](int k0) {
    // --- stage A tile (16 rows x 32 K) into LDS via im2col gather ---
    #pragma unroll 4
    for (int j = 0; j < 16; ++j) {
      int k = k0 + khalf + j;
      unsigned short bits = 0;
      if (rowok && k < K) {
        int ic = k / KK;
        int rem = k - ic * KK;
        int kh = rem / KW, kw = rem - kh * KW;
        int ih = ih0 + kh, iw = iw0 + kw;
        if ((unsigned)ih < (unsigned)H && (unsigned)iw < (unsigned)W)
          bits = in[((size_t)(n_img * C + ic) * H + ih) * W + iw];
      }
      ldsA[srow * 32 + khalf + j] = bits;
    }

    // --- async-copy B tile (32 rows x 32 K bf16) global -> LDS ---
    #pragma unroll
    for (int jj = 0; jj < 4; ++jj) {
      int row = brow_st + jj * 8;  // jj*512B / 64B-per-row
      unsigned long long g =
          wt_base + ((size_t)(n0 + row) * Kp + k0) * 2 + bbyte;
      unsigned l = ldsB_off + blin + jj * 512;
      asm volatile("global_load_async_to_lds_b128 %0, %1, off"
                   :: "v"(l), "v"(g) : "memory");
    }
    asm volatile("s_wait_asynccnt 0x0" ::: "memory");
    __syncthreads();

    // --- A fragment: two contiguous 16B LDS reads per lane ---
    v8bf alo = *(const v8bf*)&ldsA[frow * 32 + fk];
    v8bf ahi = *(const v8bf*)&ldsA[frow * 32 + fk + 16];
    v16bf a = __builtin_shufflevector(alo, ahi, 0, 1, 2, 3, 4, 5, 6, 7,
                                      8, 9, 10, 11, 12, 13, 14, 15);
    // --- B fragments for the two 16-wide N sub-tiles ---
    v8bf b0lo = *(const v8bf*)&ldsB[frow * 32 + fk];
    v8bf b0hi = *(const v8bf*)&ldsB[frow * 32 + fk + 16];
    v16bf b0 = __builtin_shufflevector(b0lo, b0hi, 0, 1, 2, 3, 4, 5, 6, 7,
                                       8, 9, 10, 11, 12, 13, 14, 15);
    v8bf b1lo = *(const v8bf*)&ldsB[(frow + 16) * 32 + fk];
    v8bf b1hi = *(const v8bf*)&ldsB[(frow + 16) * 32 + fk + 16];
    v16bf b1 = __builtin_shufflevector(b1lo, b1hi, 0, 1, 2, 3, 4, 5, 6, 7,
                                       8, 9, 10, 11, 12, 13, 14, 15);

    acc0 = __builtin_amdgcn_wmma_f32_16x16x32_bf16(false, a, false, b0,
                                                   (short)0, acc0, false, false);
    acc1 = __builtin_amdgcn_wmma_f32_16x16x32_bf16(false, a, false, b1,
                                                   (short)0, acc1, false, false);
    __syncthreads();
  };

  int k0 = 0;
  for (; k0 + 64 <= K; k0 += 64) {  // unroll x2: two chunks in flight per iter
    chunk(k0);
    chunk(k0 + 32);
  }
  for (; k0 < K; k0 += 32) chunk(k0);

  // --- epilogue: BN scale/bias + residual + ReLU, bf16 store ---
  #pragma unroll
  for (int t = 0; t < 2; ++t) {
    const v8f acc = t ? acc1 : acc0;
    const int ncol = n0 + t * 16 + (lane & 15);
    const float sc = scale ? scale[ncol] : 1.0f;
    const float bi = bias ? bias[ncol] : 0.0f;
    #pragma unroll
    for (int i = 0; i < 8; ++i) {
      int mm = m0 + i + (lane >> 4) * 8;
      if (mm >= M) continue;
      int ni = mm / (OH * OW);
      int r = mm % (OH * OW);
      int o_h = r / OW, o_w = r % OW;
      size_t oidx = ((size_t)(ni * OC + ncol) * OH + o_h) * OW + o_w;
      float v = acc[i] * sc + bi;
      if (res) v += bfbits2f(res[oidx]);
      if (relu) v = fmaxf(v, 0.0f);
      out[oidx] = f2bfbits(v);
    }
  }
}

// ---------------------------------------------------------------------------
// 3x3 s2 p1 maxpool (bf16)
// ---------------------------------------------------------------------------
__global__ void maxpool_k(const unsigned short* __restrict__ in,
                          unsigned short* __restrict__ out,
                          int NC, int H, int W, int OH, int OW) {
  int t = blockIdx.x * blockDim.x + threadIdx.x;
  if (t >= NC * OH * OW) return;
  int ow = t % OW, oh = (t / OW) % OH, nc = t / (OW * OH);
  float m = -3.0e38f;
  for (int i = 0; i < 3; ++i)
    for (int j = 0; j < 3; ++j) {
      int ih = oh * 2 - 1 + i, iw = ow * 2 - 1 + j;
      if ((unsigned)ih < (unsigned)H && (unsigned)iw < (unsigned)W)
        m = fmaxf(m, bfbits2f(in[((size_t)nc * H + ih) * W + iw]));
    }
  out[t] = f2bfbits(m);
}

// ---------------------------------------------------------------------------
// RPN heads: 1x1 convs (32->2, 32->4) with the reference transpose(0,3,2,1)
// ---------------------------------------------------------------------------
__global__ void rpn_heads_k(const unsigned short* __restrict__ rpn,
                            const float* __restrict__ cls_w, const float* __restrict__ cls_b,
                            const float* __restrict__ reg_w, const float* __restrict__ reg_b,
                            float* __restrict__ scores, float* __restrict__ deltas, int B) {
  int t = blockIdx.x * blockDim.x + threadIdx.x;
  if (t >= B * 196 * 6) return;
  int c = t % 6, i = (t / 6) % 196, b = t / (6 * 196);
  int w = i / 14, h = i % 14;  // (B,W,H,C) flatten
  if (c < 2) {
    float acc = cls_b[c];
    for (int ci = 0; ci < 32; ++ci)
      acc += bfbits2f(rpn[((size_t)(b * 32 + ci) * 14 + h) * 14 + w]) * cls_w[c * 32 + ci];
    scores[(size_t)(b * 196 + i) * 2 + c] = acc;
  } else {
    int cc = c - 2;
    float acc = reg_b[cc];
    for (int ci = 0; ci < 32; ++ci)
      acc += bfbits2f(rpn[((size_t)(b * 32 + ci) * 14 + h) * 14 + w]) * reg_w[cc * 32 + ci];
    deltas[(size_t)(b * 196 + i) * 4 + cc] = acc;
  }
}

__device__ __forceinline__ void anchor4(int i, float& a0, float& a1, float& a2, float& a3) {
  int yi = i / 14, xi = i % 14;
  float xc = 8.0f + xi * 16.0f, yc = 8.0f + yi * 16.0f;
  a0 = fmaxf(xc - 17.0f, 0.0f);
  a1 = fmaxf(yc - 17.0f, 0.0f);
  a2 = xc + 15.0f;
  a3 = yc + 15.0f;
}

__global__ void anchors_k(float* __restrict__ anch) {
  int i = blockIdx.x * blockDim.x + threadIdx.x;
  if (i >= 196) return;
  float a0, a1, a2, a3;
  anchor4(i, a0, a1, a2, a3);
  anch[i * 4 + 0] = a0; anch[i * 4 + 1] = a1;
  anch[i * 4 + 2] = a2; anch[i * 4 + 3] = a3;
}

__global__ void box_decode_k(const float* __restrict__ deltas,
                             float* __restrict__ boxes, int B) {
  int j = blockIdx.x * blockDim.x + threadIdx.x;
  if (j >= B * 196) return;
  int i = j % 196;
  float a0, a1, a2, a3;
  anchor4(i, a0, a1, a2, a3);
  float wa = a2 - a0, ha = a3 - a1;
  const float* d = deltas + (size_t)j * 4;
  float bx1 = d[0] * wa + a0;
  float by1 = d[1] * ha + a1;
  float bx2 = bx1 + expf(d[2]) * wa;
  float by2 = by1 + expf(d[3]) * ha;
  float bb[4] = {by1, bx1, by2, bx2};
  #pragma unroll
  for (int k = 0; k < 4; ++k)
    boxes[(size_t)j * 4 + k] = floorf(fminf(fmaxf(bb[k], 0.0f), 223.0f)) * (1.0f / 223.0f);
}

// ---------------------------------------------------------------------------
// crop_and_resize: fmap (B,256,14,14) bf16 + boxes -> crops (3136,256,7,7) bf16
// ---------------------------------------------------------------------------
__global__ void crop_resize_k(const unsigned short* __restrict__ fmap,
                              const float* __restrict__ boxes,
                              unsigned short* __restrict__ crops, int total) {
  int t = blockIdx.x * blockDim.x + threadIdx.x;
  if (t >= total) return;
  int p = t % 49, c = (t / 49) & 255, roi = t / (49 * 256);
  int iy = p / 7, ix = p % 7;
  int b = roi / 196;
  const float* bx = boxes + (size_t)roi * 4;
  float y1 = bx[0], x1 = bx[1], y2 = bx[2], x2 = bx[3];
  float ys = y1 * 13.0f + iy * (y2 - y1) * (13.0f / 6.0f);
  float xs = x1 * 13.0f + ix * (x2 - x1) * (13.0f / 6.0f);
  int y0 = (int)floorf(ys), yc = (int)ceilf(ys);
  int x0 = (int)floorf(xs), xc = (int)ceilf(xs);
  float ly = ys - (float)y0, lx = xs - (float)x0;
  const unsigned short* f = fmap + (size_t)(b * 256 + c) * 196;
  float f00 = bfbits2f(f[y0 * 14 + x0]);
  float f10 = bfbits2f(f[yc * 14 + x0]);
  float f01 = bfbits2f(f[y0 * 14 + xc]);
  float f11 = bfbits2f(f[yc * 14 + xc]);
  float v = (f00 * (1.0f - ly) + f10 * ly) * (1.0f - lx) +
            (f01 * (1.0f - ly) + f11 * ly) * lx;
  crops[t] = f2bfbits(v);
}

// ---------------------------------------------------------------------------
// Fused mask tail: 2x2 s2 deconv (exactly one valid tap per output pixel with
// lhs_dilation=2, pad 1) + ReLU + 1x1 conv (256->2) + sigmoid.
// One block per (roi, y, x); 256 threads = deconv output channels.
// ---------------------------------------------------------------------------
__global__ __launch_bounds__(256)
void mask_fuse_k(const unsigned short* __restrict__ m,
                 const float* __restrict__ mdw,   // (256,256,2,2)
                 const float* __restrict__ wf,    // (2,256)
                 float* __restrict__ masks) {
  __shared__ float sm[256], r0[256], r1[256];
  int blk = blockIdx.x;
  int roi = blk / 196, p = blk % 196;
  int y = p / 14, x = p % 14;
  int kh = (y + 1) & 1, kw = (x + 1) & 1;
  int py = (y - 1 + kh) >> 1, px = (x - 1 + kw) >> 1;
  int co = threadIdx.x;

  sm[co] = bfbits2f(m[((size_t)roi * 256 + co) * 49 + py * 7 + px]);
  __syncthreads();

  // w_t[co,ci,kh,kw] = w[ci,co,1-kh,1-kw]
  const int woff = co * 4 + (1 - kh) * 2 + (1 - kw);
  float acc = 0.0f;
  #pragma unroll 8
  for (int ci = 0; ci < 256; ++ci)
    acc += sm[ci] * mdw[(size_t)ci * 1024 + woff];
  float r = fmaxf(acc, 0.0f);
  r0[co] = r * wf[co];
  r1[co] = r * wf[256 + co];
  __syncthreads();
  for (int s = 128; s > 0; s >>= 1) {
    if (co < s) { r0[co] += r0[co + s]; r1[co] += r1[co + s]; }
    __syncthreads();
  }
  if (co == 0) {
    masks[((size_t)roi * 2 + 0) * 196 + p] = 1.0f / (1.0f + expf(-r0[0]));
    masks[((size_t)roi * 2 + 1) * 196 + p] = 1.0f / (1.0f + expf(-r1[0]));
  }
}

// ---------------------------------------------------------------------------
// Host-side orchestration
// ---------------------------------------------------------------------------
extern "C" void kernel_launch(void* const* d_in, const int* in_sizes, int n_in,
                              void* d_out, int out_size, void* d_ws, size_t ws_size,
                              hipStream_t stream) {
  (void)in_sizes; (void)out_size; (void)ws_size;
  if (n_in < 85 || !d_ws) return;

  const int B = 16;
  char* ws = (char*)d_ws;
  size_t off = 0;
  auto alloc = [&](size_t bytes) -> char* {
    char* p = ws + off;
    off = (off + bytes + 255) & ~(size_t)255;
    return p;
  };

  auto fin = [&](int i) { return (const float*)d_in[i]; };

  // Convert a conv weight [OC][K] f32 -> [OC][Kp] bf16 (K padded to mult of 32)
  auto prep_w = [&](int idx, int OC, int K) -> unsigned short* {
    int Kp = (K + 31) & ~31;
    unsigned short* dst = (unsigned short*)alloc((size_t)OC * Kp * 2);
    int n = OC * Kp;
    convert_w_k<<<(n + 255) / 256, 256, 0, stream>>>(fin(idx), dst, OC, K, Kp);
    return dst;
  };
  // Fold BN params (g,b,m,v at idx..idx+3) into scale/bias; returns scale, bias=+C
  auto prep_bn = [&](int idx, int C) -> float* {
    float* sb = (float*)alloc((size_t)2 * C * sizeof(float));
    fold_bn_k<<<(C + 255) / 256, 256, 0, stream>>>(fin(idx), fin(idx + 1),
                                                   fin(idx + 2), fin(idx + 3),
                                                   sb, sb + C, C);
    return sb;
  };
  auto conv = [&](const unsigned short* inp, const unsigned short* wt,
                  const float* scale, const float* bias, const unsigned short* res,
                  unsigned short* outp, int Nimg, int C, int H, int W, int OC,
                  int KH, int KW, int stride, int pad, int relu) {
    int OH = (H + 2 * pad - KH) / stride + 1;
    int OW = (W + 2 * pad - KW) / stride + 1;
    int K = C * KH * KW, Kp = (K + 31) & ~31;
    int M = Nimg * OH * OW;
    dim3 g((M + 15) / 16, OC / 32);   // wave computes 16M x 32N
    conv_wmma_k<<<g, 32, 0, stream>>>(inp, wt, scale, bias, res, outp,
                                      Nimg, C, H, W, OC, OH, OW,
                                      KH, KW, stride, pad, K, Kp, relu);
  };

  // ---- weight / BN prep (input indices per setup_inputs insertion order) ----
  unsigned short* w_conv1  = prep_w(1, 64, 3 * 49);   float* sb_bn1    = prep_bn(2, 64);
  unsigned short* w_l1b1c1 = prep_w(6, 64, 576);      float* sb_l1b1b1 = prep_bn(7, 64);
  unsigned short* w_l1b1c2 = prep_w(11, 64, 576);     float* sb_l1b1b2 = prep_bn(12, 64);
  unsigned short* w_l1b2c1 = prep_w(16, 64, 576);     float* sb_l1b2b1 = prep_bn(17, 64);
  unsigned short* w_l1b2c2 = prep_w(21, 64, 576);     float* sb_l1b2b2 = prep_bn(22, 64);
  unsigned short* w_l2b1c1 = prep_w(26, 128, 576);    float* sb_l2b1b1 = prep_bn(27, 128);
  unsigned short* w_l2b1c2 = prep_w(31, 128, 1152);   float* sb_l2b1b2 = prep_bn(32, 128);
  unsigned short* w_l2b1dc = prep_w(36, 128, 64);     float* sb_l2b1db = prep_bn(37, 128);
  unsigned short* w_l2b2c1 = prep_w(41, 128, 1152);   float* sb_l2b2b1 = prep_bn(42, 128);
  unsigned short* w_l2b2c2 = prep_w(46, 128, 1152);   float* sb_l2b2b2 = prep_bn(47, 128);
  unsigned short* w_l3b1c1 = prep_w(51, 256, 1152);   float* sb_l3b1b1 = prep_bn(52, 256);
  unsigned short* w_l3b1c2 = prep_w(56, 256, 2304);   float* sb_l3b1b2 = prep_bn(57, 256);
  unsigned short* w_l3b1dc = prep_w(61, 256, 128);    float* sb_l3b1db = prep_bn(62, 256);
  unsigned short* w_l3b2c1 = prep_w(66, 256, 2304);   float* sb_l3b2b1 = prep_bn(67, 256);
  unsigned short* w_l3b2c2 = prep_w(71, 256, 2304);   float* sb_l3b2b2 = prep_bn(72, 256);
  unsigned short* w_rpn    = prep_w(76, 32, 2304);
  unsigned short* w_mconv1 = prep_w(82, 256, 2304);

  // ---- activation buffers (bf16) ----
  const size_t xN = (size_t)B * 3 * 224 * 224;        // 2,408,448
  const size_t actN = (size_t)B * 64 * 112 * 112;     // 12,845,056 (max act)
  const size_t cropN = (size_t)3136 * 256 * 49;       // 39,337,984
  unsigned short* xbf   = (unsigned short*)alloc(xN * 2);
  unsigned short* bufA  = (unsigned short*)alloc(actN * 2);
  unsigned short* bufB  = (unsigned short*)alloc(actN * 2);
  unsigned short* bufC  = (unsigned short*)alloc(actN * 2);
  unsigned short* crops = (unsigned short*)alloc(cropN * 2);
  unsigned short* mout  = (unsigned short*)alloc(cropN * 2);
  float* boxes = (float*)alloc((size_t)3136 * 4 * sizeof(float));

  float* out_scores = (float*)d_out;            // (16,196,2)
  float* out_deltas = out_scores + 6272;        // (16,196,4)
  float* out_anch   = out_scores + 18816;       // (196,4)
  float* out_masks  = out_scores + 19600;       // (16,196,2,14,14)

  // ---- backbone ----
  cvt_f32_bf16_k<<<(int)((xN + 255) / 256), 256, 0, stream>>>(fin(0), xbf, (int)xN);

  conv(xbf, w_conv1, sb_bn1, sb_bn1 + 64, nullptr, bufA, B, 3, 224, 224, 64, 7, 7, 2, 3, 1);
  {
    int n = B * 64 * 56 * 56;
    maxpool_k<<<(n + 255) / 256, 256, 0, stream>>>(bufA, bufB, B * 64, 112, 112, 56, 56);
  }
  // layer1 (64ch, 56x56)
  conv(bufB, w_l1b1c1, sb_l1b1b1, sb_l1b1b1 + 64, nullptr, bufA, B, 64, 56, 56, 64, 3, 3, 1, 1, 1);
  conv(bufA, w_l1b1c2, sb_l1b1b2, sb_l1b1b2 + 64, bufB,    bufC, B, 64, 56, 56, 64, 3, 3, 1, 1, 1);
  conv(bufC, w_l1b2c1, sb_l1b2b1, sb_l1b2b1 + 64, nullptr, bufA, B, 64, 56, 56, 64, 3, 3, 1, 1, 1);
  conv(bufA, w_l1b2c2, sb_l1b2b2, sb_l1b2b2 + 64, bufC,    bufB, B, 64, 56, 56, 64, 3, 3, 1, 1, 1);
  // layer2 (128ch, 28x28)
  conv(bufB, w_l2b1dc, sb_l2b1db, sb_l2b1db + 128, nullptr, bufC, B, 64, 56, 56, 128, 1, 1, 2, 0, 0);
  conv(bufB, w_l2b1c1, sb_l2b1b1, sb_l2b1b1 + 128, nullptr, bufA, B, 64, 56, 56, 128, 3, 3, 2, 1, 1);
  conv(bufA, w_l2b1c2, sb_l2b1b2, sb_l2b1b2 + 128, bufC,    bufB, B, 128, 28, 28, 128, 3, 3, 1, 1, 1);
  conv(bufB, w_l2b2c1, sb_l2b2b1, sb_l2b2b1 + 128, nullptr, bufA, B, 128, 28, 28, 128, 3, 3, 1, 1, 1);
  conv(bufA, w_l2b2c2, sb_l2b2b2, sb_l2b2b2 + 128, bufB,    bufC, B, 128, 28, 28, 128, 3, 3, 1, 1, 1);
  // layer3 (256ch, 14x14)
  conv(bufC, w_l3b1dc, sb_l3b1db, sb_l3b1db + 256, nullptr, bufA, B, 128, 28, 28, 256, 1, 1, 2, 0, 0);
  conv(bufC, w_l3b1c1, sb_l3b1b1, sb_l3b1b1 + 256, nullptr, bufB, B, 128, 28, 28, 256, 3, 3, 2, 1, 1);
  conv(bufB, w_l3b1c2, sb_l3b1b2, sb_l3b1b2 + 256, bufA,    bufC, B, 256, 14, 14, 256, 3, 3, 1, 1, 1);
  conv(bufC, w_l3b2c1, sb_l3b2b1, sb_l3b2b1 + 256, nullptr, bufA, B, 256, 14, 14, 256, 3, 3, 1, 1, 1);
  conv(bufA, w_l3b2c2, sb_l3b2b2, sb_l3b2b2 + 256, bufC,    bufB, B, 256, 14, 14, 256, 3, 3, 1, 1, 1);
  // bufB = fmap (16,256,14,14)

  // ---- RPN + heads ----
  conv(bufB, w_rpn, nullptr, fin(77), nullptr, bufA, B, 256, 14, 14, 32, 3, 3, 1, 1, 1);
  {
    int n = B * 196 * 6;
    rpn_heads_k<<<(n + 255) / 256, 256, 0, stream>>>(bufA, fin(78), fin(79),
                                                     fin(80), fin(81),
                                                     out_scores, out_deltas, B);
  }
  anchors_k<<<1, 256, 0, stream>>>(out_anch);
  box_decode_k<<<(B * 196 + 255) / 256, 256, 0, stream>>>(out_deltas, boxes, B);

  // ---- ROI crops + mask head ----
  crop_resize_k<<<(int)((cropN + 255) / 256), 256, 0, stream>>>(bufB, boxes, crops, (int)cropN);
  conv(crops, w_mconv1, nullptr, nullptr, nullptr, mout, 3136, 256, 7, 7, 256, 3, 3, 1, 1, 1);
  mask_fuse_k<<<3136 * 196, 256, 0, stream>>>(mout, fin(83), fin(84), out_masks);
}
